// SDPAWithRoPEGQA_71416716198233
// MI455X (gfx1250) — matrix operand order repaired and checked by reading, not compiled, not run on hardware
//
#include <hip/hip_runtime.h>
#include <hip/hip_bf16.h>

// ---------------------------------------------------------------------------
// SDPA + RoPE + GQA for gfx1250 (MI455X), wave32, WMMA f16 + TDM path.
//   B=2, L=2048, D=2048, H=32, KVH=8, HD=64, G=4
// Pipeline:
//   1) gemm_wmma<fp32 A -> f16 out>:  Q = x@Wq, K = x@Wk, V = x@Wv  (f16 ws)
//   2) rope_kernel on Q (32 heads) and K (8 heads)
//   3) flash_attn: causal online-softmax attention, all matmuls via
//      v_wmma_f32_16x16x32_f16; Q/K tiles staged by the Tensor Data Mover
//      (tensor_load_to_lds with LDS padding), V transposed manually
//   4) gemm_wmma<f16 A -> fp32 out>:  out = O@Wo   (A tile staged by TDM)
// ---------------------------------------------------------------------------

typedef __attribute__((ext_vector_type(16))) _Float16 v16h;
typedef __attribute__((ext_vector_type(8)))  _Float16 v8h;
typedef __attribute__((ext_vector_type(8)))  float    v8f;
typedef __attribute__((ext_vector_type(4)))  float    v4f;

#define BATCH     2
#define SEQ       2048
#define DMODEL    2048
#define NH        32
#define NKV       8
#define HD        64
#define MROWS     (BATCH * SEQ)          // 4096
#define QWIDTH    (NH * HD)              // 2048
#define KWIDTH    (NKV * HD)             // 512

// ------------------------- Tensor Data Mover ---------------------------
#if defined(__has_builtin)
#if __has_builtin(__builtin_amdgcn_tensor_load_to_lds) && \
    __has_builtin(__builtin_amdgcn_s_wait_tensorcnt)
#define HAVE_TDM 1
#endif
#endif

#ifdef HAVE_TDM
typedef __attribute__((ext_vector_type(4))) unsigned int u32x4;
typedef __attribute__((ext_vector_type(8))) int          i32x8;
typedef __attribute__((ext_vector_type(4))) int          i32x4;

// 2D tile of 16-bit elements, global -> LDS, with LDS row padding.
//   width/rows      : tensor dims (elements)
//   rowStride       : tensor dim0 stride (elements)
//   tileW/tileH     : tile dims (elements)
//   padIntervalCode : 3 => every 16 DWORDs, 4 => every 32 DWORDs
//   padAmountCode   : 3 => 4 DWORDs (8 halves = LPAD)
static __device__ inline void tdm_load_2d(unsigned int ldsByteAddr,
                                          const void* gptr,
                                          unsigned int width, unsigned int rows,
                                          unsigned int rowStride,
                                          unsigned int tileW, unsigned int tileH,
                                          unsigned int padIntervalCode,
                                          unsigned int padAmountCode) {
  const unsigned long long ga = (unsigned long long)(size_t)gptr;
  u32x4 g0;
  g0[0] = 1u;                                              // count=1, no gather
  g0[1] = ldsByteAddr;                                     // lds_addr
  g0[2] = (unsigned int)(ga & 0xffffffffu);                // global_addr lo
  g0[3] = (unsigned int)((ga >> 32) & 0x1ffffffu) | (2u << 30);  // hi | type=2
  i32x8 g1;
  g1[0] = (int)((1u << 16) |                               // data_size = 2B
                (1u << 20) |                               // pad_enable
                (padIntervalCode << 22) | (padAmountCode << 25));
  g1[1] = (int)((width & 0xffffu) << 16);                  // tensor_dim0 lo
  g1[2] = (int)((width >> 16) | ((rows & 0xffffu) << 16)); // dim0 hi | dim1 lo
  g1[3] = (int)((rows >> 16) | (tileW << 16));             // dim1 hi | tile_dim0
  g1[4] = (int)(tileH & 0xffffu);                          // tile_dim1 (tile_dim2=0)
  g1[5] = (int)rowStride;                                  // tensor_dim0_stride lo
  g1[6] = 0;                                               // stride hi | dim1_stride lo
  g1[7] = 0;
  const i32x4 z4 = {0, 0, 0, 0};
  const i32x8 z8 = {0, 0, 0, 0, 0, 0, 0, 0};
  // amdgpu-toolchain (clang-23) 6-arg form: (g0, g1, g2, g3, g4, cpol)
  __builtin_amdgcn_tensor_load_to_lds(g0, g1, z4, z4, z8, 0);
}
#endif

// ------------------------------ GEMM -----------------------------------
// Block tile: 64 (M) x 128 (N), K-step 32. 256 threads = 8 waves laid out
// 2 (M) x 4 (N); each wave owns a 32x32 output tile = 2x2 wmma tiles.
#define MT 64
#define NT 128
#define KT 32
#define LPAD 8
#define ASTR (KT + LPAD)                 // 40 halves = 80B (16B multiple)

static __device__ inline v16h join16(v8h lo, v8h hi) {
  v16h r;
#pragma unroll
  for (int e = 0; e < 8; ++e) { r[e] = lo[e]; r[8 + e] = hi[e]; }
  return r;
}

template <bool A_F16, bool OUT_F16>
__global__ __launch_bounds__(256) void gemm_wmma(
    const void* __restrict__ Aptr, const float* __restrict__ W,
    void* __restrict__ Out, int M, int N, int K) {
  __shared__ __align__(16) _Float16 sA[MT * ASTR];
  __shared__ __align__(16) _Float16 sB[NT * ASTR];   // transposed: [n][k]

  const int tid  = threadIdx.x;
  const int lane = tid & 31;
  const int wave = tid >> 5;            // 0..7
  const int wm   = wave & 1;            // wave row (M)
  const int wn   = wave >> 1;           // wave col (N)
  const int lr   = lane & 15;           // lane within half-group
  const int hs   = lane >> 4;           // 0 = lanes 0-15, 1 = lanes 16-31
  const int blockM = blockIdx.y * MT;
  const int blockN = blockIdx.x * NT;

  v8f acc[2][2] = {};

  for (int k0 = 0; k0 < K; k0 += KT) {
    // ---- stage A tile (64 x 32) into LDS as f16, row-major ----
    if constexpr (A_F16) {
#ifdef HAVE_TDM
      // Tensor Data Mover: 64x32-half tile, LDS pad 4 DWORDs / 16 DWORDs.
      if (wave == 0)
        tdm_load_2d((unsigned int)(size_t)&sA[0],
                    (const _Float16*)Aptr + (size_t)blockM * K + k0,
                    (unsigned int)K, (unsigned int)M, (unsigned int)K,
                    KT, MT, /*padInterval 16dw*/ 3, /*padAmount 4dw*/ 3);
#else
      {
        const int r  = tid >> 2;
        const int cg = (tid & 3) * 8;
        const _Float16* A = (const _Float16*)Aptr;
        *(v8h*)&sA[r * ASTR + cg] =
            *(const v8h*)(A + (size_t)(blockM + r) * K + k0 + cg);
      }
#endif
    } else {
      const int r  = tid >> 2;          // 0..63
      const int cg = (tid & 3) * 8;     // 0,8,16,24
      const float* A = (const float*)Aptr;
      const float* src = A + (size_t)(blockM + r) * K + k0 + cg;
      v4f f0 = *(const v4f*)src;
      v4f f1 = *(const v4f*)(src + 4);
      v8h h;
#pragma unroll
      for (int e = 0; e < 4; ++e) {
        h[e]     = (_Float16)f0[e];
        h[4 + e] = (_Float16)f1[e];
      }
      *(v8h*)&sA[r * ASTR + cg] = h;
    }
    // ---- stage B tile (32 x 128) into LDS transposed [n][k], f16 ----
#pragma unroll
    for (int i = 0; i < 4; ++i) {
      const int chunk = tid + i * 256;  // 0..1023
      const int k  = chunk >> 5;        // 0..31
      const int ng = (chunk & 31) * 4;  // 0..124
      v4f f = *(const v4f*)(W + (size_t)(k0 + k) * N + blockN + ng);
#pragma unroll
      for (int e = 0; e < 4; ++e)
        sB[(ng + e) * ASTR + k] = (_Float16)f[e];
    }
    if (k0 + KT < K)
      __builtin_prefetch(W + (size_t)(k0 + KT) * N + blockN + (tid & 63) * 16, 0, 1);
#ifdef HAVE_TDM
    if constexpr (A_F16) {
      if (wave == 0) __builtin_amdgcn_s_wait_tensorcnt(0);
    }
#endif
    __syncthreads();

    // ---- 2x2 wmma tiles per wave, K = 32 (one wmma per tile) ----
#pragma unroll
    for (int i = 0; i < 2; ++i) {
      // A fragment: row = lane%16; lanes 0-15 hold K 0-15; 16-31 hold 16-31
      const _Float16* ab = &sA[(wm * 32 + i * 16 + lr) * ASTR];
      v16h a = join16(*(const v8h*)(ab + hs * 16),
                      *(const v8h*)(ab + hs * 16 + 8));
#pragma unroll
      for (int j = 0; j < 2; ++j) {
        // B fragment: col = lane%16; lanes 0-15 K 0-15, lanes 16-31 K 16-31
        const _Float16* bb = &sB[(wn * 32 + j * 16 + lr) * ASTR + hs * 16];
        v16h b = join16(*(const v8h*)bb, *(const v8h*)(bb + 8));
        acc[i][j] = __builtin_amdgcn_wmma_f32_16x16x32_f16(
            false, a, false, b, (short)0, acc[i][j], false, false);
      }
    }
    __syncthreads();
  }

  // ---- epilogue: D layout = lanes 0-15 rows r, lanes 16-31 rows r+8 ----
#pragma unroll
  for (int i = 0; i < 2; ++i) {
#pragma unroll
    for (int j = 0; j < 2; ++j) {
      const int col = blockN + wn * 32 + j * 16 + lr;
#pragma unroll
      for (int r = 0; r < 8; ++r) {
        const int row = blockM + wm * 32 + i * 16 + hs * 8 + r;
        if constexpr (OUT_F16)
          ((_Float16*)Out)[(size_t)row * N + col] = (_Float16)acc[i][j][r];
        else
          ((float*)Out)[(size_t)row * N + col] = acc[i][j][r];
      }
    }
  }
}

// ------------------------------ RoPE -----------------------------------
__global__ __launch_bounds__(256) void rope_kernel(_Float16* qk, int nHeads,
                                                   int rowWidth) {
  const int idx = blockIdx.x * blockDim.x + threadIdx.x;
  const int total = MROWS * nHeads * (HD / 2);
  if (idx >= total) return;
  const int i   = idx & 31;                 // 0..31
  const int h   = (idx >> 5) % nHeads;
  const int row = idx / (32 * nHeads);
  const int pos = row & (SEQ - 1);
  const float ang = (float)pos * __expf(-(float)(2 * i) * (9.2103403720f / 64.f));
  float s, c;
  __sincosf(ang, &s, &c);
  _Float16* p = qk + (size_t)row * rowWidth + h * HD;
  const float x1 = (float)p[i];
  const float x2 = (float)p[i + 32];
  p[i]      = (_Float16)(x1 * c - x2 * s);
  p[i + 32] = (_Float16)(x2 * c + x1 * s);
}

// --------------------------- Flash attention ---------------------------
// Block = (64 query rows, one head, one batch). 128 threads = 4 waves,
// each wave owns a 16-row strip. Online softmax, causal, GQA (G=4).
#define QSTR (HD + LPAD)                  // 72 halves = 144B

__global__ __launch_bounds__(128) void flash_attn(
    const _Float16* __restrict__ Q, const _Float16* __restrict__ Km,
    const _Float16* __restrict__ Vm, _Float16* __restrict__ O) {
  __shared__ __align__(16) _Float16 sQ[64 * QSTR];
  __shared__ __align__(16) _Float16 sK[64 * QSTR];       // [key][hd]
  __shared__ __align__(16) _Float16 sV[HD * QSTR];       // transposed: [hd][key]
  __shared__ __align__(16) _Float16 sP[4][16 * QSTR];    // per-wave P strip

  const int tid  = threadIdx.x;
  const int lane = tid & 31;
  const int wave = tid >> 5;              // 0..3
  const int lr   = lane & 15;
  const int hs   = lane >> 4;
  const int qt   = blockIdx.x;            // 0..31 (query tile)
  const int bh   = blockIdx.y;            // 0..63
  const int b    = bh >> 5;
  const int h    = bh & 31;
  const int kvh  = h >> 2;                // GQA: head -> kv head

  // ---- load Q tile (64 x 64 f16) ----
#ifdef HAVE_TDM
  if (wave == 0)
    tdm_load_2d((unsigned int)(size_t)&sQ[0],
                Q + (size_t)(b * SEQ + qt * 64) * QWIDTH + h * HD,
                QWIDTH, MROWS, QWIDTH, HD, 64,
                /*padInterval 32dw*/ 4, /*padAmount 4dw*/ 3);
#else
#pragma unroll
  for (int c = 0; c < 4; ++c) {
    const int chunk = tid + c * 128;      // 0..511
    const int r  = chunk >> 3;
    const int cg = (chunk & 7) * 8;
    *(v8h*)&sQ[r * QSTR + cg] =
        *(const v8h*)(Q + (size_t)(b * SEQ + qt * 64 + r) * QWIDTH + h * HD + cg);
  }
#endif

  v8f oA[4] = {};
  float mrow[8], lrow[8];
#pragma unroll
  for (int r = 0; r < 8; ++r) { mrow[r] = -1e30f; lrow[r] = 0.f; }

  for (int kb = 0; kb <= qt; ++kb) {
    __syncthreads();                      // protect sK/sV reuse
    // ---- stage K tile [key][hd] (TDM) and V tile transposed [hd][key] ----
#ifdef HAVE_TDM
    if (wave == 0)
      tdm_load_2d((unsigned int)(size_t)&sK[0],
                  Km + (size_t)(b * SEQ + kb * 64) * KWIDTH + kvh * HD,
                  KWIDTH, MROWS, KWIDTH, HD, 64, 4, 3);
#endif
#pragma unroll
    for (int c = 0; c < 4; ++c) {
      const int chunk = tid + c * 128;
      const int r  = chunk >> 3;
      const int cg = (chunk & 7) * 8;
      const size_t srow = (size_t)(b * SEQ + kb * 64 + r) * KWIDTH + kvh * HD + cg;
#ifndef HAVE_TDM
      *(v8h*)&sK[r * QSTR + cg] = *(const v8h*)(Km + srow);
#endif
      v8h v = *(const v8h*)(Vm + srow);
#pragma unroll
      for (int e = 0; e < 8; ++e) sV[(cg + e) * QSTR + r] = v[e];
    }
#ifdef HAVE_TDM
    if (wave == 0) __builtin_amdgcn_s_wait_tensorcnt(0);  // covers Q + K
#endif
    __syncthreads();

    // ---- S strip = Q[16x64] x K^T[64x64] : 8 wmma ----
    v8f s[4] = {};
#pragma unroll
    for (int kk = 0; kk < HD; kk += 32) {
      const _Float16* ab = &sQ[(wave * 16 + lr) * QSTR + kk];
      v16h a = join16(*(const v8h*)(ab + hs * 16),
                      *(const v8h*)(ab + hs * 16 + 8));
#pragma unroll
      for (int j = 0; j < 4; ++j) {
        const _Float16* bb = &sK[(j * 16 + lr) * QSTR + kk + hs * 16];
        v16h bf = join16(*(const v8h*)bb, *(const v8h*)(bb + 8));
        s[j] = __builtin_amdgcn_wmma_f32_16x16x32_f16(
            false, a, false, bf, (short)0, s[j], false, false);
      }
    }

    // ---- causal mask + scale + online softmax ----
    const int rowBase = qt * 64 + wave * 16 + hs * 8;
#pragma unroll
    for (int j = 0; j < 4; ++j) {
      const int krow = kb * 64 + j * 16 + lr;
#pragma unroll
      for (int r = 0; r < 8; ++r) {
        const float vv = s[j][r] * 0.125f;           // 1/sqrt(64)
        s[j][r] = (krow <= rowBase + r) ? vv : -1e30f;
      }
    }
#pragma unroll
    for (int r = 0; r < 8; ++r) {
      float mx = fmaxf(fmaxf(s[0][r], s[1][r]), fmaxf(s[2][r], s[3][r]));
#pragma unroll
      for (int off = 8; off >= 1; off >>= 1)         // stays in 16-lane half
        mx = fmaxf(mx, __shfl_xor(mx, off, 32));
      const float mnew = fmaxf(mrow[r], mx);
      float sum = 0.f;
#pragma unroll
      for (int j = 0; j < 4; ++j) {
        const float e = __expf(s[j][r] - mnew);
        s[j][r] = e;
        sum += e;
      }
#pragma unroll
      for (int off = 8; off >= 1; off >>= 1)
        sum += __shfl_xor(sum, off, 32);
      const float alpha = __expf(mrow[r] - mnew);
      lrow[r] = lrow[r] * alpha + sum;
      mrow[r] = mnew;
#pragma unroll
      for (int j2 = 0; j2 < 4; ++j2) oA[j2][r] *= alpha;
    }

    // ---- stage P strip through wave-private LDS (D-layout -> A-layout) ----
#pragma unroll
    for (int j = 0; j < 4; ++j)
#pragma unroll
      for (int r = 0; r < 8; ++r)
        sP[wave][(hs * 8 + r) * QSTR + j * 16 + lr] = (_Float16)s[j][r];

    // ---- O strip += P[16x64] x V[64x64] : 8 wmma ----
#pragma unroll
    for (int kk = 0; kk < 64; kk += 32) {
      const _Float16* pb = &sP[wave][lr * QSTR + kk];
      v16h aP = join16(*(const v8h*)(pb + hs * 16),
                       *(const v8h*)(pb + hs * 16 + 8));
#pragma unroll
      for (int j2 = 0; j2 < 4; ++j2) {
        const _Float16* vb = &sV[(j2 * 16 + lr) * QSTR + kk + hs * 16];
        v16h bf = join16(*(const v8h*)vb, *(const v8h*)(vb + 8));
        oA[j2] = __builtin_amdgcn_wmma_f32_16x16x32_f16(
            false, aP, false, bf, (short)0, oA[j2], false, false);
      }
    }
  }

  // ---- normalize + store O strip as f16 ----
#pragma unroll
  for (int j2 = 0; j2 < 4; ++j2) {
    const int col = h * HD + j2 * 16 + lr;
#pragma unroll
    for (int r = 0; r < 8; ++r) {
      const int row = b * SEQ + qt * 64 + wave * 16 + hs * 8 + r;
      O[(size_t)row * QWIDTH + col] = (_Float16)(oA[j2][r] / lrow[r]);
    }
  }
}

// ----------------------------- launcher --------------------------------
extern "C" void kernel_launch(void* const* d_in, const int* in_sizes, int n_in,
                              void* d_out, int out_size, void* d_ws,
                              size_t ws_size, hipStream_t stream) {
  (void)in_sizes; (void)n_in; (void)out_size; (void)ws_size;
  const float* x  = (const float*)d_in[0];
  const float* Wq = (const float*)d_in[1];
  const float* Wk = (const float*)d_in[2];
  const float* Wv = (const float*)d_in[3];
  const float* Wo = (const float*)d_in[4];
  float* out = (float*)d_out;

  char* ws = (char*)d_ws;
  _Float16* qws = (_Float16*)ws;                                   // 4096x2048
  _Float16* kws = (_Float16*)(ws + (size_t)MROWS * QWIDTH * 2);    // 4096x512
  _Float16* vws = kws + (size_t)MROWS * KWIDTH;                    // 4096x512
  _Float16* ows = vws + (size_t)MROWS * KWIDTH;                    // 4096x2048

  // 1) QKV projections (fp32 in, f16 out)
  gemm_wmma<false, true><<<dim3(QWIDTH / NT, MROWS / MT), 256, 0, stream>>>(
      x, Wq, qws, MROWS, QWIDTH, DMODEL);
  gemm_wmma<false, true><<<dim3(KWIDTH / NT, MROWS / MT), 256, 0, stream>>>(
      x, Wk, kws, MROWS, KWIDTH, DMODEL);
  gemm_wmma<false, true><<<dim3(KWIDTH / NT, MROWS / MT), 256, 0, stream>>>(
      x, Wv, vws, MROWS, KWIDTH, DMODEL);

  // 2) RoPE on Q and K
  {
    int tq = MROWS * NH * (HD / 2);
    rope_kernel<<<(tq + 255) / 256, 256, 0, stream>>>(qws, NH, QWIDTH);
    int tk = MROWS * NKV * (HD / 2);
    rope_kernel<<<(tk + 255) / 256, 256, 0, stream>>>(kws, NKV, KWIDTH);
  }

  // 3) causal flash attention with GQA
  flash_attn<<<dim3(SEQ / 64, BATCH * NH), 128, 0, stream>>>(qws, kws, vws, ows);

  // 4) output projection (f16 in, fp32 out)
  gemm_wmma<true, false><<<dim3(DMODEL / NT, MROWS / MT), 256, 0, stream>>>(
      ows, Wo, out, MROWS, DMODEL, DMODEL);
}
